// LocalPoolPointNet_39582418600282
// MI455X (gfx1250) — matrix operand order repaired and checked by reading, not compiled, not run on hardware
//
#include <hip/hip_runtime.h>

// ---------------------------------------------------------------------------
// LocalPoolPointNet encoder for MI455X (gfx1250), BF16 WMMA pipeline.
// One wave32 owns a 16-point row tile; all GEMMs are v_wmma_f32_16x16x32_bf16.
// Voxel max-pool uses monotone-u32 atomicMax (deterministic); final mean
// scatter uses float atomicAdd.
// ---------------------------------------------------------------------------

#define DEV __device__ __forceinline__

typedef __attribute__((ext_vector_type(16))) __bf16        v16bf;
typedef __attribute__((ext_vector_type(8)))  float         v8f;
typedef __attribute__((ext_vector_type(4)))  unsigned int  u32x4;

constexpr int BATCH = 2;
constexpr int NPTS  = 100000;
constexpr int H     = 128;       // hidden
constexpr int CCH   = 32;        // output channels
constexpr int RES   = 64;
constexpr int VOX   = RES * RES * RES;   // 262144
constexpr int NB    = 5;
constexpr int NT16  = NPTS / 16;         // 6250 row tiles per batch
constexpr int TILES = BATCH * NT16;      // 12500

// ---------------- scalar helpers ----------------
DEV unsigned f2u(float f) { union { float f; unsigned u; } x; x.f = f; return x.u; }
DEV float    u2f(unsigned u) { union { unsigned u; float f; } x; x.u = u; return x.f; }
DEV unsigned short f2bf(float f) {            // RNE f32 -> bf16 (finite inputs)
  unsigned u = f2u(f);
  unsigned r = (u + 0x7fffu + ((u >> 16) & 1u)) >> 16;
  return (unsigned short)r;
}
DEV float bf2f(unsigned short s) { return u2f(((unsigned)s) << 16); }
// monotone float<->u32 encoding so integer max == float max
DEV unsigned encmax(float f) { unsigned b = f2u(f); return (b & 0x80000000u) ? ~b : (b | 0x80000000u); }
DEV float    decmax(unsigned e) { unsigned b = (e & 0x80000000u) ? (e & 0x7fffffffu) : ~e; return u2f(b); }

// ---------------- WMMA fragment helpers ----------------
// A (16x32 bf16): lane<16 -> M=lane, K in {kt*32+[0..8), kt*32+16+[0..8)};
// lane>=16 -> M=lane-16, K bases shifted by +8.  Comps 0..7 = first run,
// comps 8..15 = second run (VGPR order per ISA table).
DEV v16bf afrag_row(const unsigned short* rowp, int kt, int lane) {
  const int kb = (lane >> 4) * 8;
  const unsigned short* p = rowp + kt * 32 + kb;
  union { v16bf v; u32x4 q[2]; } u;
  u.q[0] = *(const u32x4*)(p);
  u.q[1] = *(const u32x4*)(p + 16);
  return u.v;
}
DEV v16bf afrag_lds(const unsigned short* s, int rowStride, int kt, int lane) {
  const int M = lane & 15, kb = (lane >> 4) * 8;
  const unsigned short* p = s + M * rowStride + kt * 32 + kb;
  union { v16bf v; u32x4 q[2]; } u;
  u.q[0] = *(const u32x4*)(p);
  u.q[1] = *(const u32x4*)(p + 16);
  return u.v;
}
// B (32x16 bf16) pre-swizzled per (nt,kt): lane%16 = N column, 16 contiguous
// bf16 per lane = K values in the same order as the A fragment.
DEV v16bf bfrag_g(const unsigned short* w, int nt, int kt, int KT, int lane) {
  const unsigned short* p = w + ((size_t)((nt * KT + kt) * 32 + lane)) * 16;
  union { v16bf v; u32x4 q[2]; } u;
  u.q[0] = *(const u32x4*)(p);
  u.q[1] = *(const u32x4*)(p + 8);
  return u.v;
}
// gather a K=32 A-fragment from a u32-encoded pool grid row
DEV v16bf gfrag(const unsigned int* g, int lane) {
  const int kb = (lane >> 4) * 8;
  union { v16bf v; unsigned short s[16]; } u;
#pragma unroll
  for (int j = 0; j < 8; ++j) {
    u.s[j]     = f2bf(decmax(g[kb + j]));
    u.s[8 + j] = f2bf(decmax(g[kb + 16 + j]));
  }
  return u.v;
}
DEV v16bf relu16(v16bf x) {
  union { v16bf v; unsigned short s[16]; } u; u.v = x;
#pragma unroll
  for (int i = 0; i < 16; ++i) u.s[i] = (u.s[i] & 0x8000u) ? (unsigned short)0 : u.s[i];
  return u.v;
}
DEV v8f wmma_bf(v16bf a, v16bf b, v8f c) {
  return __builtin_amdgcn_wmma_f32_16x16x32_bf16(false, a, false, b, (short)0, c, false, false);
}
// C/D layout: comp r, lane<16 -> (M=r, N=lane); lane>=16 -> (M=r+8, N=lane-16)
DEV void cstore_lds(unsigned short* s, int rowStride, int nt, v8f a, int lane, bool do_relu) {
  const int Mh = (lane >> 4) * 8;
  const int Nc = nt * 16 + (lane & 15);
#pragma unroll
  for (int r = 0; r < 8; ++r) {
    float f = a[r];
    if (do_relu) f = f > 0.f ? f : 0.f;
    s[(r + Mh) * rowStride + Nc] = f2bf(f);
  }
}

// ---------------- ResNet block core (input K=256 -> out 16x128 f32) --------
DEV void resblock(const v16bf* xf,
                  const unsigned short* w0s, const float* b0,
                  const unsigned short* w1s, const float* b1,
                  const unsigned short* wss,
                  unsigned short* scr, int lane, v8f* out) {
  const int nlo = lane & 15;
  v16bf rf[8];
#pragma unroll
  for (int k = 0; k < 8; ++k) rf[k] = relu16(xf[k]);
  // h = W0 @ relu(x) + b0 ; store relu(h) to LDS for layout transpose
#pragma unroll
  for (int nt = 0; nt < 8; ++nt) {
    float bias = b0[nt * 16 + nlo];
    v8f acc;
#pragma unroll
    for (int r = 0; r < 8; ++r) acc[r] = bias;
#pragma unroll
    for (int kt = 0; kt < 8; ++kt) acc = wmma_bf(rf[kt], bfrag_g(w0s, nt, kt, 8, lane), acc);
    cstore_lds(scr, 128, nt, acc, lane, true);
  }
  v16bf hf[4];
#pragma unroll
  for (int kt = 0; kt < 4; ++kt) hf[kt] = afrag_lds(scr, 128, kt, lane);
  // out = W1 @ relu(h) + b1 + Ws @ x
#pragma unroll
  for (int nt = 0; nt < 8; ++nt) {
    float bias = b1[nt * 16 + nlo];
    v8f acc;
#pragma unroll
    for (int r = 0; r < 8; ++r) acc[r] = bias;
#pragma unroll
    for (int kt = 0; kt < 4; ++kt) acc = wmma_bf(hf[kt], bfrag_g(w1s, nt, kt, 4, lane), acc);
#pragma unroll
    for (int kt = 0; kt < 8; ++kt) acc = wmma_bf(xf[kt], bfrag_g(wss, nt, kt, 8, lane), acc);
    out[nt] = acc;
  }
}

// ---------------- fused MLP kernel (MODE 0: fc_pos+block0, 1: mid, 2: last+fc_c)
template <int MODE>
__global__ __launch_bounds__(256) void k_mlp(
    const unsigned short* __restrict__ xpk,
    unsigned short* __restrict__ net,
    const int* __restrict__ vidx,
    const unsigned int* __restrict__ pgrid,
    const unsigned short* __restrict__ wposs, const float* __restrict__ bpos,
    const unsigned short* __restrict__ w0s, const float* __restrict__ b0,
    const unsigned short* __restrict__ w1s, const float* __restrict__ b1,
    const unsigned short* __restrict__ wss,
    const unsigned short* __restrict__ wcs, const float* __restrict__ bc,
    float* __restrict__ gsum) {
  __shared__ __align__(16) unsigned short scr_all[8][16 * 256];
  const int lane = threadIdx.x & 31;
  const int wave = threadIdx.x >> 5;
  const int tile = blockIdx.x * 8 + wave;
  if (tile >= TILES) return;                 // whole-wave uniform exit
  unsigned short* scr = scr_all[wave];
  const int b    = tile / NT16;
  const int row0 = (tile - b * NT16) * 16;
  const int arow = row0 + (lane & 15);
  const int nlo  = lane & 15;

  __builtin_prefetch(w0s, 0, 0);            // global_prefetch_b8 (weights -> L2/WGP$)

  v16bf xf[8];
  if constexpr (MODE == 0) {
    // fc_pos: A is packed input (K padded 6->32), 16 output tiles of 16
    v16bf a1 = afrag_row(xpk + ((size_t)(b * NPTS + arow)) * 32, 0, lane);
#pragma unroll
    for (int nt = 0; nt < 16; ++nt) {
      float bias = bpos[nt * 16 + nlo];
      v8f acc;
#pragma unroll
      for (int r = 0; r < 8; ++r) acc[r] = bias;
      acc = wmma_bf(a1, bfrag_g(wposs, nt, 0, 1, lane), acc);
      cstore_lds(scr, 256, nt, acc, lane, false);
    }
#pragma unroll
    for (int kt = 0; kt < 8; ++kt) xf[kt] = afrag_lds(scr, 256, kt, lane);
  } else {
    const int vox = vidx[b * NPTS + arow];
    const unsigned short* nrow = net + ((size_t)(b * NPTS + arow)) * H;
#pragma unroll
    for (int kt = 0; kt < 4; ++kt) xf[kt] = afrag_row(nrow, kt, lane);
    const unsigned int* grow = pgrid + ((size_t)(b * VOX) + vox) * H;
#pragma unroll
    for (int kt = 0; kt < 4; ++kt) xf[4 + kt] = gfrag(grow + kt * 32, lane);
  }

  v8f out[8];
  resblock(xf, w0s, b0, w1s, b1, wss, scr, lane, out);

  if constexpr (MODE <= 1) {
    // bounce through LDS, then coalesced 16B stores of the 16x128 bf16 tile
#pragma unroll
    for (int nt = 0; nt < 8; ++nt) cstore_lds(scr, 128, nt, out[nt], lane, false);
    u32x4* d4 = (u32x4*)(net + ((size_t)(b * NPTS + row0)) * H);
    const u32x4* s4 = (const u32x4*)scr;
#pragma unroll
    for (int j = 0; j < 8; ++j) d4[j * 32 + lane] = s4[j * 32 + lane];
  } else {
    // fc_c fused: transpose net4 via LDS, 8 more WMMAs, scatter-add means
#pragma unroll
    for (int nt = 0; nt < 8; ++nt) cstore_lds(scr, 128, nt, out[nt], lane, false);
    v16bf hf[4];
#pragma unroll
    for (int kt = 0; kt < 4; ++kt) hf[kt] = afrag_lds(scr, 128, kt, lane);
#pragma unroll
    for (int nt = 0; nt < 2; ++nt) {
      float bias = bc[nt * 16 + nlo];
      v8f acc;
#pragma unroll
      for (int r = 0; r < 8; ++r) acc[r] = bias;
#pragma unroll
      for (int kt = 0; kt < 4; ++kt) acc = wmma_bf(hf[kt], bfrag_g(wcs, nt, kt, 4, lane), acc);
      const int Mh = (lane >> 4) * 8;
#pragma unroll
      for (int r = 0; r < 8; ++r) {
        const int M = r + Mh;
        const int v = vidx[b * NPTS + row0 + M];
        atomicAdd(&gsum[((size_t)(b * VOX) + v) * CCH + nt * 16 + nlo], acc[r]);
      }
    }
  }
}

// ---------------- small kernels ----------------
__global__ void k_zero4(u32x4* p, unsigned n4) {
  unsigned i = blockIdx.x * 256 + threadIdx.x;
  if (i < n4) { u32x4 z = {0u, 0u, 0u, 0u}; p[i] = z; }
}

__global__ __launch_bounds__(256) void k_stats(const float* __restrict__ xyz, float* __restrict__ stats) {
  __shared__ float red[256 * 6];
  const int b = blockIdx.x;
  float s[3] = {0.f, 0.f, 0.f}, q[3] = {0.f, 0.f, 0.f};
  for (int n = threadIdx.x; n < NPTS; n += 256) {
#pragma unroll
    for (int c = 0; c < 3; ++c) {
      float v = xyz[((size_t)b * NPTS + n) * 3 + c];
      s[c] += v; q[c] += v * v;
    }
  }
#pragma unroll
  for (int c = 0; c < 3; ++c) { red[threadIdx.x * 6 + c] = s[c]; red[threadIdx.x * 6 + 3 + c] = q[c]; }
  __syncthreads();
  for (int off = 128; off > 0; off >>= 1) {
    if ((int)threadIdx.x < off)
#pragma unroll
      for (int j = 0; j < 6; ++j) red[threadIdx.x * 6 + j] += red[(threadIdx.x + off) * 6 + j];
    __syncthreads();
  }
  if (threadIdx.x == 0) {
#pragma unroll
    for (int c = 0; c < 3; ++c) {
      float sum = red[c], sq = red[3 + c];
      stats[b * 3 + c]     = sum / (float)NPTS;                            // mean
      stats[6 + b * 3 + c] = (sq - sum * sum / (float)NPTS) / (float)(NPTS - 1); // var (ddof=1)
    }
  }
}

__global__ void k_fin(float* stats) {
  if (threadIdx.x == 0) {
    float m = 0.f;
#pragma unroll
    for (int j = 0; j < 6; ++j) m = fmaxf(m, stats[6 + j]);
    stats[12] = 1.0f / sqrtf(m);   // 1/deno
  }
}

__global__ __launch_bounds__(256) void k_prep(const float* __restrict__ xyz, const float* __restrict__ fea,
                                              const float* __restrict__ stats,
                                              int* __restrict__ vidx, unsigned short* __restrict__ xpk,
                                              unsigned int* __restrict__ cnt) {
  const int t = blockIdx.x * 256 + threadIdx.x;
  if (t >= BATCH * NPTS) return;
  const int b = t / NPTS;
  const float invd = stats[12];
  int xi[3];
  unsigned short* o = xpk + (size_t)t * 32;
#pragma unroll
  for (int c = 0; c < 3; ++c) {
    float v = xyz[(size_t)t * 3 + c];
    float g = (v + 0.5f) * (float)RES;
    int i = (int)g; i = i < 0 ? 0 : (i > RES - 1 ? RES - 1 : i);
    xi[c] = i;
    o[c]     = f2bf((v - stats[b * 3 + c]) * invd);
    o[3 + c] = f2bf(fea[(size_t)t * 3 + c]);
  }
#pragma unroll
  for (int c = 6; c < 32; ++c) o[c] = 0;
  const int id = xi[0] + RES * (xi[1] + RES * xi[2]);
  vidx[t] = id;
  atomicAdd(&cnt[(size_t)b * VOX + id], 1u);
}

// convert f32 weights (out,in) -> bf16 B-fragment swizzle [(nt*KT+kt)*32+lane][16 K]
__global__ void k_swz(const float* __restrict__ w, unsigned short* __restrict__ dst,
                      int nIn, int KT, int total) {
  const int t = blockIdx.x * 256 + threadIdx.x;
  if (t >= total) return;
  const int c  = t & 15;
  const int l  = (t >> 4) & 31;
  const int q  = t >> 9;           // nt*KT + kt
  const int kt = q % KT;
  const int nt = q / KT;
  const int K  = kt * 32 + (l >> 4) * 8 + (c < 8 ? c : 8 + c);
  float v = (K < nIn) ? w[(size_t)(nt * 16 + (l & 15)) * nIn + K] : 0.f;
  dst[t] = f2bf(v);
}

__global__ __launch_bounds__(256) void k_scatter(const unsigned short* __restrict__ net,
                                                 const int* __restrict__ vidx,
                                                 unsigned int* __restrict__ pgrid) {
  const int t = blockIdx.x * 256 + threadIdx.x;
  if (t >= BATCH * NPTS * 32) return;
  const int p = t >> 5;           // point
  const int g = t & 31;           // channel group (4 ch)
  const int b = p / NPTS;
  const unsigned short* nr = net + (size_t)p * H + g * 4;
  unsigned int* grow = pgrid + ((size_t)(b * VOX) + vidx[p]) * H + g * 4;
#pragma unroll
  for (int j = 0; j < 4; ++j) atomicMax(&grow[j], encmax(bf2f(nr[j])));
}

__global__ void k_norm(const float* __restrict__ gsum, const unsigned int* __restrict__ cnt,
                       float* __restrict__ out) {
  const unsigned t = blockIdx.x * 256 + threadIdx.x;
  if (t >= (unsigned)(BATCH * CCH * VOX)) return;
  const unsigned b = t / (CCH * VOX);
  const unsigned r = t - b * (CCH * VOX);
  const unsigned c = r / VOX;
  const unsigned v = r - c * VOX;
  const unsigned ct = cnt[(size_t)b * VOX + v];
  out[t] = ct ? gsum[((size_t)b * VOX + v) * CCH + c] / (float)ct : 0.f;
}

// ---------------------------------------------------------------------------
extern "C" void kernel_launch(void* const* d_in, const int* in_sizes, int n_in,
                              void* d_out, int out_size, void* d_ws, size_t ws_size,
                              hipStream_t stream) {
  // inputs in setup_inputs() dict (insertion) order
  const float* xyz  = (const float*)d_in[0];
  const float* fea  = (const float*)d_in[1];
  const float* wposf = (const float*)d_in[2];
  const float* bpos  = (const float*)d_in[3];
  const float *w0f[NB], *b0f[NB], *w1f[NB], *b1f[NB], *wsf[NB];
  for (int i = 0; i < NB; ++i) {
    w0f[i] = (const float*)d_in[4 + 5 * i];
    b0f[i] = (const float*)d_in[5 + 5 * i];
    w1f[i] = (const float*)d_in[6 + 5 * i];
    b1f[i] = (const float*)d_in[7 + 5 * i];
    wsf[i] = (const float*)d_in[8 + 5 * i];
  }
  const float* wcf = (const float*)d_in[29];
  const float* bcf = (const float*)d_in[30];

  // workspace carve-out (~385 MiB)
  char* base = (char*)d_ws;
  size_t off = 0;
  auto take = [&](size_t bytes) -> void* {
    void* p = base + off;
    off = (off + bytes + 255) & ~(size_t)255;
    return p;
  };
  unsigned int*   grid  = (unsigned int*)take((size_t)BATCH * VOX * H * 4);   // pool max (u32 monotone)
  float*          gsum  = (float*)take((size_t)BATCH * VOX * CCH * 4);
  unsigned int*   cnt   = (unsigned int*)take((size_t)BATCH * VOX * 4);
  unsigned short* net   = (unsigned short*)take((size_t)BATCH * NPTS * H * 2);
  unsigned short* xpk   = (unsigned short*)take((size_t)BATCH * NPTS * 32 * 2);
  int*            vidx  = (int*)take((size_t)BATCH * NPTS * 4);
  float*          stats = (float*)take(256);
  unsigned short* wposs = (unsigned short*)take(16 * 1 * 512 * 2);
  unsigned short *w0s[NB], *w1s[NB], *wss[NB];
  for (int i = 0; i < NB; ++i) {
    w0s[i] = (unsigned short*)take(8 * 8 * 512 * 2);
    w1s[i] = (unsigned short*)take(8 * 4 * 512 * 2);
    wss[i] = (unsigned short*)take(8 * 8 * 512 * 2);
  }
  unsigned short* wcs = (unsigned short*)take(2 * 4 * 512 * 2);
  (void)ws_size; (void)in_sizes; (void)n_in; (void)out_size;

  // zero accumulators + pool grid
  {
    unsigned n4g = (unsigned)((size_t)BATCH * VOX * H / 4);
    unsigned n4s = (unsigned)((size_t)BATCH * VOX * CCH / 4);
    unsigned n4c = (unsigned)((size_t)BATCH * VOX / 4);
    k_zero4<<<dim3((n4g + 255) / 256), 256, 0, stream>>>((u32x4*)grid, n4g);
    k_zero4<<<dim3((n4s + 255) / 256), 256, 0, stream>>>((u32x4*)gsum, n4s);
    k_zero4<<<dim3((n4c + 255) / 256), 256, 0, stream>>>((u32x4*)cnt, n4c);
  }
  // stats (mean/var per batch, fixed-order reduce) -> inv_deno
  k_stats<<<dim3(BATCH), 256, 0, stream>>>(xyz, stats);
  k_fin<<<dim3(1), 32, 0, stream>>>(stats);
  // per-point voxel index, packed bf16 input, voxel counts
  k_prep<<<dim3((BATCH * NPTS + 255) / 256), 256, 0, stream>>>(xyz, fea, stats, vidx, xpk, cnt);
  // weight convert + swizzle
  auto swz = [&](const float* wf, unsigned short* dst, int nOut, int nIn, int KT) {
    int total = (nOut / 16) * KT * 512;
    k_swz<<<dim3((total + 255) / 256), 256, 0, stream>>>(wf, dst, nIn, KT, total);
  };
  swz(wposf, wposs, 256, 6, 1);
  for (int i = 0; i < NB; ++i) {
    swz(w0f[i], w0s[i], 128, 256, 8);
    swz(w1f[i], w1s[i], 128, 128, 4);
    swz(wsf[i], wss[i], 128, 256, 8);
  }
  swz(wcf, wcs, 32, 128, 4);

  const dim3 gm((TILES + 7) / 8);
  const dim3 gs((BATCH * NPTS * 32 + 255) / 256);
  const unsigned n4g = (unsigned)((size_t)BATCH * VOX * H / 4);

  // fc_pos + block0 -> net0
  k_mlp<0><<<gm, 256, 0, stream>>>(xpk, net, vidx, nullptr, wposs, bpos,
                                   w0s[0], b0f[0], w1s[0], b1f[0], wss[0],
                                   nullptr, nullptr, nullptr);
  // blocks 1..4: scatter-max(net_{i-1}) -> gather + resblock
  for (int i = 1; i < NB; ++i) {
    k_scatter<<<gs, 256, 0, stream>>>(net, vidx, grid);
    if (i < NB - 1) {
      k_mlp<1><<<gm, 256, 0, stream>>>(nullptr, net, vidx, grid, nullptr, nullptr,
                                       w0s[i], b0f[i], w1s[i], b1f[i], wss[i],
                                       nullptr, nullptr, nullptr);
      k_zero4<<<dim3((n4g + 255) / 256), 256, 0, stream>>>((u32x4*)grid, n4g);
    } else {
      k_mlp<2><<<gm, 256, 0, stream>>>(nullptr, net, vidx, grid, nullptr, nullptr,
                                       w0s[i], b0f[i], w1s[i], b1f[i], wss[i],
                                       wcs, bcf, gsum);
    }
  }
  // per-voxel mean -> [B, C, R, R, R]
  k_norm<<<dim3((unsigned)(BATCH * CCH * VOX + 255) / 256), 256, 0, stream>>>(gsum, cnt, (float*)d_out);
}
